// DKL_70970039599698
// MI455X (gfx1250) — matrix operand order repaired and checked by reading, not compiled, not run on hardware
//
#include <hip/hip_runtime.h>
#include <hip/hip_bf16.h>

typedef _Float16 half_t;
typedef __attribute__((ext_vector_type(16))) _Float16 v16h;
typedef __attribute__((ext_vector_type(8)))  _Float16 v8h;
typedef __attribute__((ext_vector_type(8)))  float    v8f;
typedef __attribute__((address_space(3))) half_t* lds_hptr_t;

#define N_PTS   4096
#define D_IN    1024
#define D_FEAT  128
#define TILE    128
#define BSTR    40     // LDS row stride (halves) for 32-wide K staging (pad vs bank conflicts)
#define FSTR    136    // LDS row stride (halves) for feature column block

// ---------------------------------------------------------------------------
// WMMA fragment loaders, per CDNA5 ISA 7.12.2 (16-bit A 16x32, B 32x16, wave32)
// A: lane<16 -> M=lane, halves = [K k0..k0+7] ++ [K k0+16..k0+23]
//    lane>=16 -> M=lane-16, halves = [K k0+8..k0+15] ++ [K k0+24..k0+31]
// B[K][N] = rows[N][K]: lane<16 -> N=lane, halves = [K k0..k0+7] ++ [K k0+16..k0+23]
//           lane>=16 -> N=lane-16, halves = [K k0+8..k0+15] ++ [K k0+24..k0+31]
// ---------------------------------------------------------------------------
__device__ __forceinline__ v16h frag_A(const half_t* __restrict__ p, int ld, int k0) {
  const int lane = threadIdx.x & 31;
  const int m    = lane & 15;
  const int hi   = lane >> 4;
  const half_t* q = p + (size_t)m * ld + k0 + hi * 8;
  v8h lo = *(const v8h*)(q);
  v8h hh = *(const v8h*)(q + 16);
  return __builtin_shufflevector(lo, hh, 0,1,2,3,4,5,6,7,8,9,10,11,12,13,14,15);
}

__device__ __forceinline__ v16h frag_B(const half_t* __restrict__ p, int ld, int k0) {
  const int lane = threadIdx.x & 31;
  const int n    = lane & 15;
  const int hi   = lane >> 4;
  const half_t* q = p + (size_t)n * ld + k0 + hi * 8;
  v8h lo = *(const v8h*)(q);
  v8h hh = *(const v8h*)(q + 16);
  return __builtin_shufflevector(lo, hh, 0,1,2,3,4,5,6,7,8,9,10,11,12,13,14,15);
}

__device__ __forceinline__ float wave_sum(float v) {
  v += __shfl_xor(v, 16, 32);
  v += __shfl_xor(v,  8, 32);
  v += __shfl_xor(v,  4, 32);
  v += __shfl_xor(v,  2, 32);
  v += __shfl_xor(v,  1, 32);
  return v;
}
__device__ __forceinline__ float half_sum(float v) {  // reduce within 16-lane halves
  v += __shfl_xor(v, 8, 32);
  v += __shfl_xor(v, 4, 32);
  v += __shfl_xor(v, 2, 32);
  v += __shfl_xor(v, 1, 32);
  return v;
}

// ---------------------------------------------------------------------------
// Utility kernels
// ---------------------------------------------------------------------------
__global__ void zero_kernel(float* __restrict__ p, int n) {
  int i = blockIdx.x * blockDim.x + threadIdx.x;
  if (i < n) p[i] = 0.0f;
}

__global__ void cvt_kernel(const float* __restrict__ s, half_t* __restrict__ d, int n) {
  int i = blockIdx.x * blockDim.x + threadIdx.x;
  if (i < n) d[i] = (half_t)s[i];
}

// W (K x N, row-major) -> Wt16 (N x K, row-major f16)
__global__ void cvtT_kernel(const float* __restrict__ W, half_t* __restrict__ Wt) {
  int i = blockIdx.x * blockDim.x + threadIdx.x;
  if (i < D_IN * D_FEAT) {
    int k = i / D_FEAT, n = i % D_FEAT;
    Wt[(size_t)n * D_IN + k] = (half_t)W[i];
  }
}

// fp32 exact row squared-norms; one wave per row
__global__ __launch_bounds__(256) void rownorm_kernel(const float* __restrict__ src,
                                                      float* __restrict__ dst, int K) {
  int row  = blockIdx.x * 8 + (threadIdx.x >> 5);
  int lane = threadIdx.x & 31;
  const float* p = src + (size_t)row * K;
  float s = 0.0f;
  for (int k = lane; k < K; k += 32) { float v = p[k]; s += v * v; }
  s = wave_sum(s);
  if (lane == 0) dst[row] = s;
}

// ---------------------------------------------------------------------------
// F = Y @ W + b  via WMMA  (M=4096, N=128, K=1024); stores fp32 and f16
// ---------------------------------------------------------------------------
__global__ __launch_bounds__(256) void feat_kernel(const half_t* __restrict__ Y16,
                                                   const half_t* __restrict__ Wt16,
                                                   const float* __restrict__ bias,
                                                   float* __restrict__ F32,
                                                   half_t* __restrict__ F16o) {
  const int i0   = blockIdx.x * TILE;
  const int wave = threadIdx.x >> 5;
  const int lane = threadIdx.x & 31;
  const int hi   = lane >> 4;
  const int nlo  = lane & 15;

  const v8f zero8 = {0.f,0.f,0.f,0.f,0.f,0.f,0.f,0.f};
  v8f acc[8];
#pragma unroll
  for (int a = 0; a < 8; ++a) acc[a] = zero8;

  const half_t* Arow = Y16 + (size_t)(i0 + wave * 16) * D_IN;
#pragma unroll 1
  for (int k0 = 0; k0 < D_IN; k0 += 32) {
    v16h af = frag_A(Arow, D_IN, k0);
#pragma unroll
    for (int a = 0; a < 8; ++a) {
      v16h bf = frag_B(Wt16 + (size_t)(16 * a) * D_IN, D_IN, k0);
      acc[a] = __builtin_amdgcn_wmma_f32_16x16x32_f16(false, af, false, bf,
                                                      (short)0, acc[a], false, false);
    }
  }
#pragma unroll
  for (int a = 0; a < 8; ++a) {
    float bb = bias[16 * a + nlo];
#pragma unroll
    for (int r = 0; r < 8; ++r) {
      float v   = acc[a][r] + bb;
      int   row = i0 + wave * 16 + r + hi * 8;
      int   col = 16 * a + nlo;
      F32[(size_t)row * D_FEAT + col]  = v;
      F16o[(size_t)row * D_FEAT + col] = (half_t)v;
    }
  }
}

// ---------------------------------------------------------------------------
// Fused MMD tile kernel: one 128x128 tile of the 4096x4096 domain per block.
// B-chunks are streamed with GLOBAL_LOAD_ASYNC_TO_LDS_B128 (ASYNCcnt), double
// buffered so the DMA of chunk k+1 overlaps the WMMA chain on chunk k.
// Accumulates: scal[0..5] = {S_Kx, S_Ky, S_Kxy, T_Kx, T_Ky, T_Kxy}; row_hh[i].
// ---------------------------------------------------------------------------
__global__ __launch_bounds__(256) void mmd_tiles(
    const half_t* __restrict__ X16, const half_t* __restrict__ Y16,
    const half_t* __restrict__ F16,
    const float* __restrict__ xn, const float* __restrict__ yn,
    const float* __restrict__ fn,
    const float* __restrict__ epsp, const float* __restrict__ sKp,
    const float* __restrict__ sqp,
    float* __restrict__ row_hh, float* __restrict__ scal) {
  __shared__ half_t Fsh[TILE * FSTR];     // whole feature column block (128 x 128 halves)
  __shared__ half_t Bsh[2][TILE * BSTR];  // double-buffered 32-wide K chunk

  const int  bi = blockIdx.y, bj = blockIdx.x;
  const int  i0 = bi * TILE, j0 = bj * TILE;
  const int  wave = threadIdx.x >> 5;
  const int  lane = threadIdx.x & 31;
  const int  hi   = lane >> 4;
  const int  nlo  = lane & 15;
  const bool diagTile = (bi == bj);

  const float eps = *epsp;
  const float isK = 1.0f / *sKp;
  const float isq = 1.0f / *sqp;
  const float ome = 1.0f - eps;
  const v8f zero8 = {0.f,0.f,0.f,0.f,0.f,0.f,0.f,0.f};

  // LDS byte offset of the staging buffer (addrspace(3) pointer value == offset)
  const unsigned bshOff = (unsigned)(unsigned long long)(lds_hptr_t)&Bsh[0][0];
  // this thread's slice of a 128x32-half chunk: 32 bytes = 2 x b128
  const int srow = threadIdx.x >> 1;
  const int scol = (threadIdx.x & 1) * 16;                       // halves
  const unsigned ldsSlice = (unsigned)((srow * BSTR + scol) * 2);  // bytes

  // ---- stage feature column block to LDS ----
  for (int t = threadIdx.x; t < TILE * 16; t += 256) {  // 16 x uint4 per row
    int row = t >> 4, c8 = (t & 15) * 8;
    *(uint4*)&Fsh[row * FSTR + c8] =
        *(const uint4*)&F16[(size_t)(j0 + row) * D_FEAT + c8];
  }
  __syncthreads();

  // ---- Gff = F_i . F_j^T over K=128, then convert to G per element ----
  v8f g[8];
#pragma unroll
  for (int a = 0; a < 8; ++a) g[a] = zero8;
  {
    const half_t* Fa = F16 + (size_t)(i0 + wave * 16) * D_FEAT;
#pragma unroll 1
    for (int k0 = 0; k0 < D_FEAT; k0 += 32) {
      v16h af = frag_A(Fa, D_FEAT, k0);
#pragma unroll
      for (int a = 0; a < 8; ++a) {
        v16h bf = frag_B(&Fsh[16 * a * FSTR], FSTR, k0);
        g[a] = __builtin_amdgcn_wmma_f32_16x16x32_f16(false, af, false, bf,
                                                      (short)0, g[a], false, false);
      }
    }
  }
  float fr[8], fc[8];
#pragma unroll
  for (int r = 0; r < 8; ++r) fr[r] = fn[i0 + wave * 16 + r + hi * 8];
#pragma unroll
  for (int a = 0; a < 8; ++a) fc[a] = fn[j0 + 16 * a + nlo];
#pragma unroll
  for (int a = 0; a < 8; ++a) {
#pragma unroll
    for (int r = 0; r < 8; ++r) {
      float D  = fmaxf(fr[r] + fc[a] - 2.0f * g[a][r], 0.0f);
      g[a][r]  = ome * __expf(-D * isK) + eps;  // G = (1-eps)*exp(-Dfeat/sK) + eps
    }
  }

  // ---- per-lane norms for the big Grams ----
  float xr[8], yr[8], xc[8], yc[8];
#pragma unroll
  for (int r = 0; r < 8; ++r) {
    int row = i0 + wave * 16 + r + hi * 8;
    xr[r] = xn[row]; yr[r] = yn[row];
  }
#pragma unroll
  for (int a = 0; a < 8; ++a) {
    int col = j0 + 16 * a + nlo;
    xc[a] = xn[col]; yc[a] = yn[col];
  }

  v8f hh[8];
#pragma unroll
  for (int a = 0; a < 8; ++a) hh[a] = zero8;
  float sXX = 0.f, sYY = 0.f, sXY = 0.f, tXX = 0.f, tYY = 0.f, tXY = 0.f;

  // issue this thread's 32-byte slice of chunk k0 into Bsh[buf] (async -> LDS)
  auto stage_async = [&](const half_t* __restrict__ Brow, int k0, int buf) {
    unsigned long long src =
        (unsigned long long)(const void*)(Brow + (size_t)srow * D_IN + k0 + scol);
    unsigned dst = bshOff + (unsigned)(buf * (TILE * BSTR * 2)) + ldsSlice;
    asm volatile(
        "global_load_async_to_lds_b128 %0, %1, off\n\t"
        "global_load_async_to_lds_b128 %0, %1, off offset:16"
        :
        : "v"(dst), "v"(src)
        : "memory");
  };

  // K=1024 Gram with async double-buffered LDS staging of the 128x32 B chunk
  auto gram = [&](const half_t* __restrict__ Arow, const half_t* __restrict__ Brow,
                  v8f (&acc)[8]) {
#pragma unroll
    for (int a = 0; a < 8; ++a) acc[a] = zero8;
    stage_async(Brow, 0, 0);
    asm volatile("s_wait_asynccnt 0x0" ::: "memory");
    __syncthreads();
#pragma unroll 1
    for (int k0 = 0; k0 < D_IN; k0 += 32) {
      const int cur = (k0 >> 5) & 1;
      if (k0 + 32 < D_IN) stage_async(Brow, k0 + 32, cur ^ 1);  // overlap DMA w/ WMMA
      v16h af = frag_A(Arow, D_IN, k0);
#pragma unroll
      for (int a = 0; a < 8; ++a) {
        v16h bf = frag_B(&Bsh[cur][16 * a * BSTR], BSTR, 0);
        acc[a] = __builtin_amdgcn_wmma_f32_16x16x32_f16(false, af, false, bf,
                                                        (short)0, acc[a], false, false);
      }
      asm volatile("s_wait_asynccnt 0x0" ::: "memory");
      __syncthreads();  // next buffer fully landed; everyone done reading cur
    }
  };

  // fold one Gram into hh + scalar partials:  t = G * exp(-Dorg/sq)
  auto combine = [&](v8f (&acc)[8], const float* nr, const float* nc, float sign,
                     float& S, float* T) {
#pragma unroll
    for (int a = 0; a < 8; ++a) {
#pragma unroll
      for (int r = 0; r < 8; ++r) {
        float D = fmaxf(nr[r] + nc[a] - 2.0f * acc[a][r], 0.0f);
        float t = g[a][r] * __expf(-D * isq);
        S += t;
        hh[a][r] += sign * t;
        if (T != nullptr) {
          if (diagTile && a == wave && nlo == r + hi * 8) *T += t;
        }
      }
    }
  };

  v8f acc[8];
  const half_t* Xi = X16 + (size_t)(i0 + wave * 16) * D_IN;
  const half_t* Yi = Y16 + (size_t)(i0 + wave * 16) * D_IN;
  const half_t* Xj = X16 + (size_t)j0 * D_IN;
  const half_t* Yj = Y16 + (size_t)j0 * D_IN;

  gram(Xi, Xj, acc); combine(acc, xr, xc, +1.0f, sXX, &tXX);  // Kx
  gram(Yi, Yj, acc); combine(acc, yr, yc, +1.0f, sYY, &tYY);  // Ky
  gram(Xi, Yj, acc); combine(acc, xr, yc, -1.0f, sXY, &tXY);  // Kxy
  float sDum = 0.f;
  gram(Yi, Xj, acc); combine(acc, yr, xc, -1.0f, sDum, nullptr);  // Kxy^T

  // ---- scalar reductions (one atomic per wave per scalar) ----
  sXX = wave_sum(sXX); sYY = wave_sum(sYY); sXY = wave_sum(sXY);
  tXX = wave_sum(tXX); tYY = wave_sum(tYY); tXY = wave_sum(tXY);
  if (lane == 0) {
    atomicAdd(&scal[0], sXX);
    atomicAdd(&scal[1], sYY);
    atomicAdd(&scal[2], sXY);
    atomicAdd(&scal[3], tXX);
    atomicAdd(&scal[4], tYY);
    atomicAdd(&scal[5], tXY);
  }

  // ---- row sums of hh ----
#pragma unroll
  for (int r = 0; r < 8; ++r) {
    float v = 0.f;
#pragma unroll
    for (int a = 0; a < 8; ++a) v += hh[a][r];
    v = half_sum(v);
    if (nlo == 0) atomicAdd(&row_hh[i0 + wave * 16 + r + hi * 8], v);
  }
}

// ---------------------------------------------------------------------------
// Final assembly of (mmd_hat, mmd_std_hat, ratio)
// ---------------------------------------------------------------------------
__global__ __launch_bounds__(256) void final_kernel(const float* __restrict__ row_hh,
                                                    const float* __restrict__ scal,
                                                    float* __restrict__ out) {
  __shared__ float red[8];
  float s = 0.f;
  for (int i = threadIdx.x; i < N_PTS; i += 256) { float r = row_hh[i]; s += r * r; }
  s = wave_sum(s);
  if ((threadIdx.x & 31) == 0) red[threadIdx.x >> 5] = s;
  __syncthreads();
  if (threadIdx.x == 0) {
    float rowsq = 0.f;
#pragma unroll
    for (int w = 0; w < 8; ++w) rowsq += red[w];
    const float nx = (float)N_PTS, ny = (float)N_PTS;
    float SKx = scal[0], SKy = scal[1], SKxy = scal[2];
    float TKx = scal[3], TKy = scal[4], TKxy = scal[5];
    float xx = (SKx - TKx) / (nx * (nx - 1.0f));
    float yy = (SKy - TKy) / (ny * (ny - 1.0f));
    float xy = (SKxy - TKxy) / (nx * (ny - 1.0f));
    float mmd2   = xx - 2.0f * xy + yy;
    float sum_hh = SKx + SKy - 2.0f * SKxy;
    float V1 = rowsq / (ny * ny * ny);       // dot(row,row)/ny with row = rowsum/ny
    float V2 = sum_hh / (nx * nx);
    float varEst  = 4.0f * (V1 - V2 * V2);
    float mmd_hat = -mmd2;
    float stdv    = sqrtf(varEst + 1e-8f);
    out[0] = mmd_hat;
    out[1] = stdv;
    out[2] = mmd_hat / stdv;
  }
}

// ---------------------------------------------------------------------------
extern "C" void kernel_launch(void* const* d_in, const int* in_sizes, int n_in,
                              void* d_out, int out_size, void* d_ws, size_t ws_size,
                              hipStream_t stream) {
  const float* X    = (const float*)d_in[0];
  const float* Y    = (const float*)d_in[1];
  const float* W    = (const float*)d_in[2];
  const float* bias = (const float*)d_in[3];
  const float* epsp = (const float*)d_in[4];
  const float* sKp  = (const float*)d_in[5];
  const float* sqp  = (const float*)d_in[6];
  float* out = (float*)d_out;

  // workspace layout
  char*  ws = (char*)d_ws;
  size_t o  = 0;
  half_t* X16  = (half_t*)(ws + o); o += (size_t)N_PTS * D_IN * sizeof(half_t);
  half_t* Y16  = (half_t*)(ws + o); o += (size_t)N_PTS * D_IN * sizeof(half_t);
  half_t* Wt16 = (half_t*)(ws + o); o += (size_t)D_FEAT * D_IN * sizeof(half_t);
  half_t* F16  = (half_t*)(ws + o); o += (size_t)N_PTS * D_FEAT * sizeof(half_t);
  float*  F32  = (float*)(ws + o);  o += (size_t)N_PTS * D_FEAT * sizeof(float);
  float*  xn   = (float*)(ws + o);  o += (size_t)N_PTS * sizeof(float);
  float*  yn   = (float*)(ws + o);  o += (size_t)N_PTS * sizeof(float);
  float*  fn   = (float*)(ws + o);  o += (size_t)N_PTS * sizeof(float);
  float*  rowhh= (float*)(ws + o);  o += (size_t)N_PTS * sizeof(float);
  float*  scal = (float*)(ws + o);  o += 16 * sizeof(float);
  if (ws_size < o) return;  // insufficient scratch; refuse to scribble

  const int nXY = N_PTS * D_IN;

  zero_kernel<<<(N_PTS + 16 + 255) / 256, 256, 0, stream>>>(rowhh, N_PTS + 16);
  cvt_kernel<<<nXY / 256, 256, 0, stream>>>(X, X16, nXY);
  cvt_kernel<<<nXY / 256, 256, 0, stream>>>(Y, Y16, nXY);
  cvtT_kernel<<<(D_IN * D_FEAT) / 256, 256, 0, stream>>>(W, Wt16);
  rownorm_kernel<<<N_PTS / 8, 256, 0, stream>>>(X, xn, D_IN);
  rownorm_kernel<<<N_PTS / 8, 256, 0, stream>>>(Y, yn, D_IN);
  feat_kernel<<<N_PTS / TILE, 256, 0, stream>>>(Y16, Wt16, bias, F32, F16);
  rownorm_kernel<<<N_PTS / 8, 256, 0, stream>>>(F32, fn, D_FEAT);
  mmd_tiles<<<dim3(32, 32), 256, 0, stream>>>(X16, Y16, F16, xn, yn, fn,
                                              epsp, sKp, sqp, rowhh, scal);
  final_kernel<<<1, 256, 0, stream>>>(rowhh, scal, out);

  (void)in_sizes; (void)n_in; (void)out_size;
}
